// AttGRU_14061722927433
// MI455X (gfx1250) — matrix operand order repaired and checked by reading, not compiled, not run on hardware
//
#include <hip/hip_runtime.h>
#include <hip/hip_bf16.h>

// AttGRU (bidirectional AGRU), H=768, B=64, S=1024, fp32 in/out.
//
// Strategy (MI455X / gfx1250, wave32, WMMA):
//  1) k_convert : weights -> bf16 in ws.
//  2) k_proj    : hoist the time-parallel input projections out of the scan:
//                 P_r[d,t,b,n] = c[b,t,:]·Wr_dᵀ + (bWr+bUr),  P_h = c·Wᵀ + bW,
//                 big bf16 WMMA GEMMs, stored bf16 (402 MB ws).
//  3) k_scan    : persistent kernel, 16 WGs per direction, each owns 48 cols.
//                 Recurrent weight slices live in LDS (147 KB) for all 1024
//                 steps; per step each wave does 4 v_wmma tiles; h is
//                 broadcast through L2 with an atomic per-direction barrier.
// ws requirement ~413 MB.

#define Sd 1024
#define Bd 64
#define Hd 768
#define H2 (2 * Hd)
#define NWGD 16            // workgroups per direction
#define NCOL 48            // output columns per workgroup (16*48 = 768)
#define SCAN_THREADS 192   // 6 waves of 32

typedef __attribute__((ext_vector_type(16))) __bf16 v16bf;
typedef __attribute__((ext_vector_type(8)))  __bf16 v8bf;
typedef __attribute__((ext_vector_type(8)))  float  v8f;

union ABu { v16bf v; v8bf h[2]; };

// A-matrix 16x32 bf16 fragment (ISA 7.12.2): lane L -> row M = L&15,
// elems 0..7  = K[k0 + 8*half + 0..7], elems 8..15 = K[k0 + 16 + 8*half + 0..7]
__device__ __forceinline__ v16bf load_a_frag(const __bf16* S, int mt, int k0, int lane) {
  const int half = lane >> 4;
  const int m = mt * 16 + (lane & 15);
  ABu u;
  const __bf16* p = S + m * Hd + k0 + 8 * half;
  u.h[0] = *(const v8bf*)(p);
  u.h[1] = *(const v8bf*)(p + 16);
  return u.v;
}

// B-matrix 32x16 bf16 fragment: lane L -> col N = L&15,
// elems 0..7 = K[k0 + 8*half + 0..7], elems 8..15 = K[k0 + 16 + 8*half + 0..7] (row n of row-major W[n][k])
__device__ __forceinline__ v16bf load_b_frag(const __bf16* W, int nt, int k0, int lane) {
  const int half = lane >> 4;
  const int n = nt * 16 + (lane & 15);
  ABu u;
  const __bf16* p = W + (long)n * Hd + k0 + 8 * half;
  u.h[0] = *(const v8bf*)(p);
  u.h[1] = *(const v8bf*)(p + 16);
  return u.v;
}

__device__ __forceinline__ v8f wmma_bf16(v16bf a, v16bf b, v8f c) {
  return __builtin_amdgcn_wmma_f32_16x16x32_bf16(false, a, false, b, (short)0, c,
                                                 false, false);
}

// ---------------------------------------------------------------- k_convert
__global__ void k_convert(const float* w0, const float* w1, const float* w2,
                          const float* w3, const float* w4, const float* w5,
                          const float* w6, const float* w7, __bf16* dst) {
  const float* ws[8] = {w0, w1, w2, w3, w4, w5, w6, w7};
  const long N = 8L * Hd * Hd;
  for (long i = (long)blockIdx.x * blockDim.x + threadIdx.x; i < N;
       i += (long)gridDim.x * blockDim.x) {
    long s = i / (Hd * Hd);
    long e = i - s * (Hd * Hd);
    dst[i] = (__bf16)ws[s][e];
  }
}

// ---------------------------------------------------------------- k_init
__global__ void k_init(const float* h0, float* Hbuf, unsigned* bar) {
  long gid = (long)blockIdx.x * blockDim.x + threadIdx.x;
  if (gid < 2) bar[gid] = 0;                 // reset barrier counters each launch
  const long N = (long)Bd * Hd;
  if (gid < 2 * N) {                          // parity 0, both directions
    long d = gid / N, e = gid - d * N;
    Hbuf[d * N + e] = h0[e];
  }
}

// ---------------------------------------------------------------- k_proj
// One block per time step t; M=64 (batch), N=768, K=768; c staged to LDS bf16.
__global__ __launch_bounds__(256) void k_proj(const float* __restrict__ ctx,
                                              const __bf16* __restrict__ Wm,
                                              const float* __restrict__ b1,
                                              const float* __restrict__ b2,
                                              __bf16* __restrict__ P) {
  extern __shared__ char smemp[];
  __bf16* sC = (__bf16*)smemp;  // [64][768]
  const int t = blockIdx.x;
  const int tid = threadIdx.x;
  for (int idx = tid; idx < Bd * Hd; idx += 256) {
    int b = idx / Hd, k = idx - b * Hd;
    sC[idx] = (__bf16)ctx[((long)b * Sd + t) * Hd + k];
  }
  __syncthreads();
  const int wv = tid >> 5, lane = tid & 31;
  const int mt = wv & 3, ntg = wv >> 2;
  const int half = lane >> 4, nl = lane & 15;
  for (int nc = 0; nc < Hd; nc += 128) {
    v8f acc[4] = {};
    const __bf16* Wc = Wm + (long)nc * Hd;
    for (int k0 = 0; k0 < Hd; k0 += 32) {
      v16bf a = load_a_frag(sC, mt, k0, lane);
#pragma unroll
      for (int i = 0; i < 4; ++i)
        acc[i] = wmma_bf16(a, load_b_frag(Wc, ntg + 2 * i, k0, lane), acc[i]);
    }
#pragma unroll
    for (int i = 0; i < 4; ++i) {
      const int n = nc + (ntg + 2 * i) * 16 + nl;
      float bb = b1[n];
      if (b2) bb += b2[n];
#pragma unroll
      for (int v = 0; v < 8; ++v) {
        const int m = mt * 16 + v + 8 * half;
        P[((long)t * Bd + m) * Hd + n] = (__bf16)(acc[i][v] + bb);
      }
    }
  }
}

// ---------------------------------------------------------------- k_scan
__device__ __forceinline__ void dirbar(unsigned* c, unsigned target) {
  __threadfence();          // release this WG's global stores (agent scope)
  __syncthreads();
  if (threadIdx.x == 0) {
    __hip_atomic_fetch_add(c, 1u, __ATOMIC_RELEASE, __HIP_MEMORY_SCOPE_AGENT);
    while (__hip_atomic_load(c, __ATOMIC_ACQUIRE, __HIP_MEMORY_SCOPE_AGENT) < target)
      __builtin_amdgcn_s_sleep(2);
  }
  __syncthreads();
}

__device__ __forceinline__ void epilogue(v8f aUr, v8f aU, int mt, int nt, int lane,
                                         int n0, int d, int t, int tau,
                                         const __bf16* Pr, const __bf16* Ph,
                                         const float* att, const float* sBU,
                                         const float* Hread, float* Hwrite,
                                         float* out) {
  const int half = lane >> 4, nl = lane & 15;
  const int nloc = nt * 16 + nl;
  const int n = n0 + nloc;
  const float bu = sBU[nloc];
  const long pbase = (((long)d * Sd + tau) * Bd) * (long)Hd + n;
#pragma unroll
  for (int v = 0; v < 8; ++v) {
    const int m = mt * 16 + v + 8 * half;
    const float prv = (float)Pr[pbase + (long)m * Hd];
    const float phv = (float)Ph[pbase + (long)m * Hd];
    const float r = 1.0f / (1.0f + __expf(-(prv + aUr[v])));
    const float ht = tanhf(phv + r * (aU[v] + bu));
    const float g = att[m * Sd + tau];
    const float hn = g * ht + (1.0f - g) * Hread[m * Hd + n];
    Hwrite[m * Hd + n] = hn;
    if (t == Sd - 1) out[m * H2 + d * Hd + n] = hn;
  }
}

__global__ __launch_bounds__(SCAN_THREADS) void k_scan(
    const __bf16* __restrict__ Pr, const __bf16* __restrict__ Ph,
    const __bf16* __restrict__ wB, const float* __restrict__ att,
    const float* __restrict__ bU_f, const float* __restrict__ bU_b,
    float* __restrict__ Hbuf, unsigned* __restrict__ bar,
    float* __restrict__ out) {
  extern __shared__ char smem[];
  __bf16* sH = (__bf16*)smem;                                    // [64][768] bf16
  __bf16* sW = (__bf16*)(smem + 2 * Bd * Hd);                    // [2][48][768] bf16
  float* sBU = (float*)(smem + 2 * Bd * Hd + 2 * 2 * NCOL * Hd); // [48] f32

  const int tid = threadIdx.x;
  const int d = blockIdx.x >> 4;        // direction
  const int wgd = blockIdx.x & 15;      // workgroup within direction
  const int n0 = wgd * NCOL;
  const __bf16* UrG = wB + (long)(4 + 2 * d) * Hd * Hd;
  const __bf16* UG  = wB + (long)(5 + 2 * d) * Hd * Hd;
  const float* bU = d ? bU_b : bU_f;

  // LDS-resident recurrent weight slices for all 1024 steps (147 KB).
  for (int idx = tid; idx < 2 * NCOL * Hd; idx += SCAN_THREADS) {
    int sel = idx / (NCOL * Hd);
    int e = idx - sel * (NCOL * Hd);
    int r = e / Hd, k = e - r * Hd;
    const __bf16* src = sel ? UG : UrG;
    sW[idx] = src[(long)(n0 + r) * Hd + k];
  }
  if (tid < NCOL) sBU[tid] = bU[n0 + tid];
  __syncthreads();

  const int wv = tid >> 5, lane = tid & 31;
  const int nt = wv >> 1;               // 6 waves -> 3 n-tiles x {m-tiles 0-1, m-tiles 2-3}
  const int mt0 = (2 * wv) & 3;
  const int mt1 = mt0 + 1;
  unsigned* cnt = bar + d;

  for (int t = 0; t < Sd; ++t) {
    const int tau = d ? (Sd - 1 - t) : t;
    const int prty = t & 1;
    const float* Hread  = Hbuf + (long)(prty * 2 + d) * Bd * Hd;
    float*       Hwrite = Hbuf + (long)((prty ^ 1) * 2 + d) * Bd * Hd;

    // Stage h (fp32 -> bf16) into LDS.
    for (int idx = tid; idx < Bd * Hd; idx += SCAN_THREADS)
      sH[idx] = (__bf16)Hread[idx];
    // Barrier L(t): everyone done reading parity t&1 before it is rewritten.
    dirbar(cnt, (unsigned)(2 * t + 1) * NWGD);

    // Prefetch next step's projection tile into cache (global_prefetch_b8).
    if (t + 1 < Sd) {
      const int taun = d ? (Sd - 2 - t) : (t + 1);
      const int mpre = tid & 63;
      const int npre = n0 + ((tid >> 6) << 4);
      const long pb = (((long)d * Sd + taun) * Bd + mpre) * (long)Hd + npre;
      __builtin_prefetch(Pr + pb, 0, 1);
      __builtin_prefetch(Ph + pb, 0, 1);
    }

    // GEMMs: hUr = h·Urᵀ and hU = h·Uᵀ for this WG's 48 columns.
    v8f aUr0 = {}, aU0 = {}, aUr1 = {}, aU1 = {};
    for (int k0 = 0; k0 < Hd; k0 += 32) {
      v16bf a0 = load_a_frag(sH, mt0, k0, lane);
      v16bf a1 = load_a_frag(sH, mt1, k0, lane);
      v16bf bUrF = load_b_frag(sW, nt, k0, lane);
      v16bf bUF  = load_b_frag(sW + NCOL * Hd, nt, k0, lane);
      aUr0 = wmma_bf16(a0, bUrF, aUr0);
      aU0  = wmma_bf16(a0, bUF,  aU0);
      aUr1 = wmma_bf16(a1, bUrF, aUr1);
      aU1  = wmma_bf16(a1, bUF,  aU1);
    }

    epilogue(aUr0, aU0, mt0, nt, lane, n0, d, t, tau, Pr, Ph, att, sBU, Hread, Hwrite, out);
    epilogue(aUr1, aU1, mt1, nt, lane, n0, d, t, tau, Pr, Ph, att, sBU, Hread, Hwrite, out);

    // Barrier W(t): all h_new writes visible before next step reads them.
    dirbar(cnt, (unsigned)(2 * t + 2) * NWGD);
  }
}

// ---------------------------------------------------------------- host
extern "C" void kernel_launch(void* const* d_in, const int* in_sizes, int n_in,
                              void* d_out, int out_size, void* d_ws, size_t ws_size,
                              hipStream_t stream) {
  const float* ctx   = (const float*)d_in[0];
  const float* h0    = (const float*)d_in[1];
  const float* att   = (const float*)d_in[2];
  const float* Wr_f  = (const float*)d_in[3];
  const float* bWr_f = (const float*)d_in[4];
  const float* Ur_f  = (const float*)d_in[5];
  const float* bUr_f = (const float*)d_in[6];
  const float* W_f   = (const float*)d_in[7];
  const float* bW_f  = (const float*)d_in[8];
  const float* U_f   = (const float*)d_in[9];
  const float* bU_f  = (const float*)d_in[10];
  const float* Wr_b  = (const float*)d_in[11];
  const float* bWr_b = (const float*)d_in[12];
  const float* Ur_b  = (const float*)d_in[13];
  const float* bUr_b = (const float*)d_in[14];
  const float* W_b   = (const float*)d_in[15];
  const float* bW_b  = (const float*)d_in[16];
  const float* U_b   = (const float*)d_in[17];
  const float* bU_b  = (const float*)d_in[18];
  float* out = (float*)d_out;

  // ws layout (bytes): [0) 8 bf16 weight mats (9,437,184)
  //                    [.) P_r bf16 2*S*B*H  (201,326,592)
  //                    [.) P_h bf16          (201,326,592)
  //                    [.) Hbuf f32 2par*2dir*B*H (786,432)
  //                    [.) barrier counters (2x u32)
  char* ws = (char*)d_ws;
  __bf16* wB = (__bf16*)ws;
  const size_t offPr = (size_t)8 * Hd * Hd * 2;
  const size_t szP   = (size_t)2 * Sd * Bd * Hd * 2;
  __bf16* Pr = (__bf16*)(ws + offPr);
  __bf16* Ph = (__bf16*)(ws + offPr + szP);
  const size_t offH = offPr + 2 * szP;
  float* Hbuf = (float*)(ws + offH);
  unsigned* bar = (unsigned*)(ws + offH + (size_t)2 * 2 * Bd * Hd * 4);

  const int smemProj = Bd * Hd * 2;                                   // 96 KB
  const int smemScan = Bd * Hd * 2 + 2 * NCOL * Hd * 2 + NCOL * 4;    // ~240 KB
  hipFuncSetAttribute(reinterpret_cast<const void*>(k_proj),
                      hipFuncAttributeMaxDynamicSharedMemorySize, smemProj);
  hipFuncSetAttribute(reinterpret_cast<const void*>(k_scan),
                      hipFuncAttributeMaxDynamicSharedMemorySize, smemScan);

  // Slot order: 0:Wr_f 1:W_f 2:Wr_b 3:W_b 4:Ur_f 5:U_f 6:Ur_b 7:U_b
  k_convert<<<512, 256, 0, stream>>>(Wr_f, W_f, Wr_b, W_b, Ur_f, U_f, Ur_b, U_b, wB);
  k_init<<<(2 * Bd * Hd + 255) / 256, 256, 0, stream>>>(h0, Hbuf, bar);

  k_proj<<<Sd, 256, smemProj, stream>>>(ctx, wB + 0L * Hd * Hd, bWr_f, bUr_f, Pr);
  k_proj<<<Sd, 256, smemProj, stream>>>(ctx, wB + 1L * Hd * Hd, bW_f, nullptr, Ph);
  k_proj<<<Sd, 256, smemProj, stream>>>(ctx, wB + 2L * Hd * Hd, bWr_b, bUr_b,
                                        Pr + (size_t)Sd * Bd * Hd);
  k_proj<<<Sd, 256, smemProj, stream>>>(ctx, wB + 3L * Hd * Hd, bW_b, nullptr,
                                        Ph + (size_t)Sd * Bd * Hd);

  k_scan<<<2 * NWGD, SCAN_THREADS, smemScan, stream>>>(Pr, Ph, wB, att, bU_f, bU_b,
                                                       Hbuf, bar, out);
  (void)in_sizes; (void)n_in; (void)out_size; (void)ws_size;
}
